// MOLEDGL_26766236189156
// MI455X (gfx1250) — compile-verified, matrix operand order checked by reading, not run on hardware
//
#include <hip/hip_runtime.h>
#include <hip/hip_bf16.h>

// ---------------------------------------------------------------------------
// MoLE mixed-expert linear:  y = segment_mm(x, sum_e c[b,e]*W[e]) + bias
// Phase 1: mix weights -> bf16 hi/lo split in workspace.
// Phase 2: 16 batched GEMMs via v_wmma_f32_16x16x32_bf16, bf16x3 compensation.
// ---------------------------------------------------------------------------

typedef __attribute__((ext_vector_type(16))) __bf16 v16bf;
typedef __attribute__((ext_vector_type(8)))  float  v8f;

static constexpr int E    = 8;
static constexpr int I    = 1024;
static constexpr int O    = 1024;
static constexpr int Bseg = 16;
static constexpr int N    = 16384;

static constexpr int BM = 128;   // block rows
static constexpr int BN = 128;   // block cols
static constexpr int BK = 32;    // K per WMMA / per LDS tile
static constexpr int LDSS = BK + 8;  // padded LDS stride in halves (40) -> conflict-free

__device__ __forceinline__ unsigned short f32_to_bf16_rne(float f) {
  unsigned u = __float_as_uint(f);
  unsigned r = u + 0x7FFFu + ((u >> 16) & 1u);   // round to nearest even
  return (unsigned short)(r >> 16);
}
__device__ __forceinline__ float bf16_to_f32(unsigned short h) {
  return __uint_as_float((unsigned)h << 16);
}

union FragU { uint4 q[2]; v16bf v; };

// Load one 16x32 bf16 A/B fragment from an LDS tile stored [row][k] with
// stride LDSS halves.  Matches ISA 16-bit A-matrix layout:
//   lanes 0-15 : M=lane, K = 0..7 (VGPR0-3) and 16..23 (VGPR4-7)
//   lanes 16-31: M=lane-16, K = 8..15 and 24..31
__device__ __forceinline__ v16bf load_frag(const unsigned short* base, int lane) {
  const int m  = lane & 15;
  const int k0 = (lane & 16) ? 8 : 0;
  const unsigned short* p = base + m * LDSS + k0;   // 16B aligned (LDSS*2=80, k0*2=16)
  FragU u;
  u.q[0] = *(const uint4*)(p);        // K = k0 .. k0+7
  u.q[1] = *(const uint4*)(p + 16);   // K = k0+16 .. k0+23
  return u.v;
}

// ---------------------------------------------------------------------------
// Kernel 1: W_mix[b][o][i] = sum_e coeff[b][e] * W[e][o][i], split to bf16 hi/lo.
// Weights are read exactly once (loop over all segments inside the thread).
// ---------------------------------------------------------------------------
__global__ void __launch_bounds__(256) mole_mix_weights(
    const float* __restrict__ W, const float* __restrict__ coeff,
    unsigned short* __restrict__ Whi, unsigned short* __restrict__ Wlo) {
  __shared__ float c[Bseg * E];
  if (threadIdx.x < Bseg * E) c[threadIdx.x] = coeff[threadIdx.x];
  __syncthreads();

  const int unit = blockIdx.x * blockDim.x + threadIdx.x;  // O*I/4 units total
  const int o  = unit >> 8;          // I/4 = 256 units per output row
  const int i4 = (unit & 255) << 2;  // 4 consecutive i per thread

  float w[E][4];
#pragma unroll
  for (int e = 0; e < E; ++e) {
    const float4 t = *(const float4*)&W[((size_t)e * O + o) * I + i4];
    w[e][0] = t.x; w[e][1] = t.y; w[e][2] = t.z; w[e][3] = t.w;
  }

#pragma unroll
  for (int b = 0; b < Bseg; ++b) {
    float s[4] = {0.f, 0.f, 0.f, 0.f};
#pragma unroll
    for (int e = 0; e < E; ++e) {
      const float cb = c[b * E + e];
#pragma unroll
      for (int j = 0; j < 4; ++j) s[j] = fmaf(cb, w[e][j], s[j]);
    }
    ushort4 hi, lo;
    hi.x = f32_to_bf16_rne(s[0]); lo.x = f32_to_bf16_rne(s[0] - bf16_to_f32(hi.x));
    hi.y = f32_to_bf16_rne(s[1]); lo.y = f32_to_bf16_rne(s[1] - bf16_to_f32(hi.y));
    hi.z = f32_to_bf16_rne(s[2]); lo.z = f32_to_bf16_rne(s[2] - bf16_to_f32(hi.z));
    hi.w = f32_to_bf16_rne(s[3]); lo.w = f32_to_bf16_rne(s[3] - bf16_to_f32(hi.w));
    const size_t idx = ((size_t)b * O + o) * I + i4;
    *(ushort4*)&Whi[idx] = hi;
    *(ushort4*)&Wlo[idx] = lo;
  }
}

// ---------------------------------------------------------------------------
// Kernel 2: per 128x128 output tile: y = x_seg @ W_mix[seg]^T + bias.
// bf16x3: acc += aH*bH + aH*bL + aL*bH  (near-f32 accuracy, f32 accumulate).
// ---------------------------------------------------------------------------
__global__ void __launch_bounds__(256) mole_gemm(
    const float* __restrict__ x,
    const unsigned short* __restrict__ Whi,
    const unsigned short* __restrict__ Wlo,
    const float* __restrict__ bias,
    const int* __restrict__ mole_sizes,
    float* __restrict__ out) {
  __shared__ unsigned short sAh[BM * LDSS];
  __shared__ unsigned short sAl[BM * LDSS];
  __shared__ unsigned short sBh[BN * LDSS];
  __shared__ unsigned short sBl[BN * LDSS];   // 4 * 10240B = 40 KB of 320 KB

  const int tid  = threadIdx.x;
  const int lane = tid & 31;
  const int wid  = tid >> 5;                  // 8 wave32s
  const int rowStart = blockIdx.y * BM;
  const int o0       = blockIdx.x * BN;

  // Segment for this row-block (row blocks do not straddle segments here).
  int seg = 0, cum = mole_sizes[0];
  while (rowStart >= cum && seg < Bseg - 1) { ++seg; cum += mole_sizes[seg]; }

  const int wRow = (wid & 3) * 32;            // 4 waves along M
  const int wCol = (wid >> 2) * 64;           // 2 waves along N

  const v8f vzero = {};
  v8f acc[2][4];
#pragma unroll
  for (int mi = 0; mi < 2; ++mi)
#pragma unroll
    for (int ni = 0; ni < 4; ++ni) acc[mi][ni] = vzero;

  // A-tile loader mapping: 256 threads cover 128 rows x 2 half-rows of 16 floats.
  const int ar = tid >> 1;
  const int ak = (tid & 1) * 16;
  const size_t wBase = ((size_t)seg * O + o0) * I;

  for (int k0 = 0; k0 < I; k0 += BK) {
    // ---- stage A tile: f32 -> (hi,lo) bf16 into LDS ----
    const float* gx = &x[(size_t)(rowStart + ar) * I + k0 + ak];
#pragma unroll
    for (int j = 0; j < 4; ++j) {
      const float4 t = *(const float4*)(gx + 4 * j);
      ushort4 h, l;
      h.x = f32_to_bf16_rne(t.x); l.x = f32_to_bf16_rne(t.x - bf16_to_f32(h.x));
      h.y = f32_to_bf16_rne(t.y); l.y = f32_to_bf16_rne(t.y - bf16_to_f32(h.y));
      h.z = f32_to_bf16_rne(t.z); l.z = f32_to_bf16_rne(t.z - bf16_to_f32(h.z));
      h.w = f32_to_bf16_rne(t.w); l.w = f32_to_bf16_rne(t.w - bf16_to_f32(h.w));
      const int d = ar * LDSS + ak + 4 * j;
      *(ushort4*)&sAh[d] = h;
      *(ushort4*)&sAl[d] = l;
    }
    if (k0 + BK < I) __builtin_prefetch(gx + BK, 0, 3);  // global_prefetch_b8 next K tile

    // ---- stage B tile: bf16 hi/lo straight copies, [o][i] layout ----
#pragma unroll
    for (int it = 0; it < 4; ++it) {
      const int linear = it * 256 + tid;        // 1024 uint4 transfers
      const int buf = linear >> 9;              // 0 = hi, 1 = lo
      const int rem = linear & 511;
      const int r = rem >> 2;
      const int q = (rem & 3) * 8;
      const unsigned short* src =
          (buf ? Wlo : Whi) + wBase + (size_t)r * I + k0 + q;
      const uint4 t = *(const uint4*)src;
      unsigned short* dst = (buf ? sBl : sBh) + r * LDSS + q;
      *(uint4*)dst = t;
    }
    __syncthreads();

    // ---- compute: 24 wmma per wave per K-step ----
    v16bf aH[2], aL[2];
#pragma unroll
    for (int mi = 0; mi < 2; ++mi) {
      aH[mi] = load_frag(&sAh[(wRow + mi * 16) * LDSS], lane);
      aL[mi] = load_frag(&sAl[(wRow + mi * 16) * LDSS], lane);
    }
#pragma unroll
    for (int ni = 0; ni < 4; ++ni) {
      const v16bf bH = load_frag(&sBh[(wCol + ni * 16) * LDSS], lane);
      const v16bf bL = load_frag(&sBl[(wCol + ni * 16) * LDSS], lane);
#pragma unroll
      for (int mi = 0; mi < 2; ++mi) {
        acc[mi][ni] = __builtin_amdgcn_wmma_f32_16x16x32_bf16(
            false, aH[mi], false, bH, (short)0, acc[mi][ni], false, false);
        acc[mi][ni] = __builtin_amdgcn_wmma_f32_16x16x32_bf16(
            false, aH[mi], false, bL, (short)0, acc[mi][ni], false, false);
        acc[mi][ni] = __builtin_amdgcn_wmma_f32_16x16x32_bf16(
            false, aL[mi], false, bH, (short)0, acc[mi][ni], false, false);
      }
    }
    __syncthreads();
  }

  // ---- epilogue: C layout VGPR r -> (M=r lanes 0-15, M=r+8 lanes 16-31) ----
  const int nOff  = lane & 15;
  const int mHalf = (lane >> 4) * 8;
#pragma unroll
  for (int mi = 0; mi < 2; ++mi) {
#pragma unroll
    for (int ni = 0; ni < 4; ++ni) {
      const int col = o0 + wCol + ni * 16 + nOff;
      const float bv = bias[col];
#pragma unroll
      for (int r = 0; r < 8; ++r) {
        const int row = rowStart + wRow + mi * 16 + mHalf + r;
        out[(size_t)row * O + col] = acc[mi][ni][r] + bv;
      }
    }
  }
}

// ---------------------------------------------------------------------------
extern "C" void kernel_launch(void* const* d_in, const int* in_sizes, int n_in,
                              void* d_out, int out_size, void* d_ws, size_t ws_size,
                              hipStream_t stream) {
  (void)in_sizes; (void)n_in; (void)out_size; (void)ws_size;
  const float* x     = (const float*)d_in[0];   // [N, I]
  const float* W     = (const float*)d_in[1];   // [E, O, I]
  const float* bias  = (const float*)d_in[2];   // [O]
  const float* coeff = (const float*)d_in[3];   // [B, E]
  const int* msz     = (const int*)d_in[4];     // [B]
  float* out         = (float*)d_out;           // [N, O]

  unsigned short* Whi = (unsigned short*)d_ws;              // 32 MB
  unsigned short* Wlo = Whi + (size_t)Bseg * O * I;         // 32 MB

  mole_mix_weights<<<(O * I / 4) / 256, 256, 0, stream>>>(W, coeff, Whi, Wlo);
  mole_gemm<<<dim3(O / BN, N / BM), 256, 0, stream>>>(x, Whi, Wlo, bias, msz, out);
}
